// SpikingSRWKV_9234179687069
// MI455X (gfx1250) — matrix operand order repaired and checked by reading, not compiled, not run on hardware
//
#include <hip/hip_runtime.h>
#include <math.h>

// Problem constants (match reference)
#define D_      1024
#define B_      8
#define T_      1024
#define M_      (B_*T_)      // 8192 flattened rows (b*T + t)
#define NN_     64
#define DT_     0.02f
#define THR_    0.5f
#define SLOPE_  10.0f

#define NBLK_   32           // persistent blocks for recurrence (1 per WGP)
#define CPB_    (D_/NBLK_)   // 32 channels per block

typedef __bf16 bf16_t;
typedef bf16_t v16bf __attribute__((ext_vector_type(16)));
typedef float  v8f   __attribute__((ext_vector_type(8)));
typedef unsigned u32x4 __attribute__((ext_vector_type(4)));
typedef int      i32x4 __attribute__((ext_vector_type(4)));
typedef int      i32x8 __attribute__((ext_vector_type(8)));

// ---------------------------------------------------------------------------
// WMMA fragment loaders, per CDNA5 ISA 7.12.2 (wave32)
// ---------------------------------------------------------------------------
__device__ __forceinline__ v16bf load_a_frag(const bf16_t* A, int ldk,
                                             int m0, int k0, int lane) {
  const int m  = m0 + (lane & 15);
  const int kb = k0 + ((lane & 16) ? 8 : 0);
  const bf16_t* p = A + (size_t)m * ldk + kb;
  v16bf f;
#pragma unroll
  for (int e = 0; e < 8; ++e) { f[e] = p[e]; f[e + 8] = p[16 + e]; }
  return f;
}

__device__ __forceinline__ v16bf load_b_frag(const bf16_t* W, int ldk,
                                             int n0, int k0, int lane) {
  const int n  = n0 + (lane & 15);
  const int kb = k0 + ((lane & 16) ? 8 : 0);
  const bf16_t* p = W + (size_t)n * ldk + kb;
  v16bf f;
#pragma unroll
  for (int e = 0; e < 8; ++e) { f[e] = p[e]; f[e + 8] = p[16 + e]; }
  return f;
}

__device__ __forceinline__ float sigm_(float x) { return 1.0f / (1.0f + expf(-x)); }

// ---------------------------------------------------------------------------
// Tensor Data Mover: 2D tile (rows x cols bf16 elements, row stride in elems)
// from global memory into LDS. Descriptor per CDNA5 ISA sections 8.3/8.4.
// Issued once per calling wave (TDM ignores EXEC); caller restricts to wave 0.
// ---------------------------------------------------------------------------
#if __has_builtin(__builtin_amdgcn_tensor_load_to_lds)
#define HAVE_TDM_ 1
__device__ __forceinline__ void tdm_load_tile_2d(unsigned lds_addr, const void* gaddr,
                                                 unsigned rows, unsigned cols,
                                                 unsigned stride_elems) {
  const unsigned long long ga = (unsigned long long)gaddr;
  u32x4 g0;
  g0[0] = 1u;                                     // count=1, user-mode descriptor
  g0[1] = lds_addr;                               // LDS byte address
  g0[2] = (unsigned)(ga & 0xffffffffu);           // global_addr[31:0]
  g0[3] = (unsigned)((ga >> 32) & 0x01ffffffu)    // global_addr[56:32]
        | (2u << 30);                             // type = 2 ("image")
  i32x8 g1;
  g1[0] = (int)(1u << 16);                        // wg_mask=0, data_size=1 (2B)
  g1[1] = (int)((cols & 0xffffu) << 16);          // tensor_dim0[15:0]
  g1[2] = (int)((cols >> 16) & 0xffffu)           // tensor_dim0[31:16]
        | (int)((rows & 0xffffu) << 16);          // tensor_dim1[15:0]
  g1[3] = (int)((rows >> 16) & 0xffffu)           // tensor_dim1[31:16]
        | (int)((cols & 0xffffu) << 16);          // tile_dim0 = cols
  g1[4] = (int)(rows & 0xffffu);                  // tile_dim1 = rows, tile_dim2 = 0
  g1[5] = (int)stride_elems;                      // tensor_dim0_stride[31:0]
  g1[6] = 0;                                      // stride0 hi / stride1 lo
  g1[7] = 0;                                      // stride1 hi
  const i32x4 z4 = {0, 0, 0, 0};
#if defined(__clang_major__) && (__clang_major__ >= 23)
  const i32x8 z8 = {0, 0, 0, 0, 0, 0, 0, 0};
  __builtin_amdgcn_tensor_load_to_lds(g0, g1, z4, z4, z8, 0);
#else
  __builtin_amdgcn_tensor_load_to_lds(g0, g1, z4, z4, 0);
#endif
}
#else
#define HAVE_TDM_ 0
#endif

// ---------------------------------------------------------------------------
// Elementwise helpers
// ---------------------------------------------------------------------------
__global__ void cvt_kernel(const float* __restrict__ s, bf16_t* __restrict__ d, int n) {
  int i = blockIdx.x * blockDim.x + threadIdx.x;
  if (i < n) d[i] = (bf16_t)s[i];
}

__global__ void zero_kernel(unsigned* __restrict__ p, int n) {
  int i = blockIdx.x * blockDim.x + threadIdx.x;
  if (i < n) p[i] = 0u;
}

// ---------------------------------------------------------------------------
// Batched WMMA GEMM: out[m][n] = epilogue( sum_k A[m][k]*W[n][k] (+bias[n]) )
// Wave computes a 16x32 tile (2 accumulators sharing one A fragment).
// Block = nwaves*32 threads covering 16 x (nwaves*32) output.
// op 0: store DT / (0.5*sigmoid(v+bias) + 0.02) ; op 1: bias-add ; op 2: raw
// ---------------------------------------------------------------------------
__global__ void __launch_bounds__(128) wmma_gemm_bf16(
    const bf16_t* __restrict__ A, const bf16_t* __restrict__ W,
    const float* __restrict__ bias, float* __restrict__ out, int N, int op) {
  const int lane = threadIdx.x & 31;
  const int wave = threadIdx.x >> 5;
  const int m0 = blockIdx.x * 16;
  const int n0 = blockIdx.y * (blockDim.x) + wave * 32;  // blockDim.x/32 waves * 32 cols

  v8f acc0 = {0.f, 0.f, 0.f, 0.f, 0.f, 0.f, 0.f, 0.f};
  v8f acc1 = acc0;
  for (int k0 = 0; k0 < D_; k0 += 32) {
    v16bf a  = load_a_frag(A, D_, m0, k0, lane);
    v16bf b0 = load_b_frag(W, D_, n0, k0, lane);
    v16bf b1 = load_b_frag(W, D_, n0 + 16, k0, lane);
    acc0 = __builtin_amdgcn_wmma_f32_16x16x32_bf16(false, a, false, b0,
                                                   (short)0, acc0, false, false);
    acc1 = __builtin_amdgcn_wmma_f32_16x16x32_bf16(false, a, false, b1,
                                                   (short)0, acc1, false, false);
  }

#pragma unroll
  for (int half = 0; half < 2; ++half) {
    const v8f acc = half ? acc1 : acc0;
    const int nb = n0 + half * 16;
#pragma unroll
    for (int j = 0; j < 8; ++j) {
      const int m = m0 + j + ((lane & 16) ? 8 : 0);
      const int n = nb + (lane & 15);
      float v = acc[j];
      if (bias) v += bias[n];
      if (op == 0) v = DT_ / (0.5f * sigm_(v) + 0.02f);
      out[(size_t)m * N + n] = v;
    }
  }
}

// ---------------------------------------------------------------------------
// Top-k(8) over 64 router logits + softmax -> sparse gate (in-place buffer)
// ---------------------------------------------------------------------------
__global__ void topk_gate_kernel(float* __restrict__ gate, int rows) {
  int r = blockIdx.x * blockDim.x + threadIdx.x;
  if (r >= rows) return;
  float* p = gate + (size_t)r * NN_;
  float v[NN_];
#pragma unroll
  for (int i = 0; i < NN_; ++i) v[i] = p[i];
  int idx[8]; float val[8];
#pragma unroll
  for (int k = 0; k < 8; ++k) {
    int bi = 0; float bv = -3.0e38f;
    for (int i = 0; i < NN_; ++i) if (v[i] > bv) { bv = v[i]; bi = i; }
    idx[k] = bi; val[k] = bv; v[bi] = -3.0e38f;
  }
  const float mx = val[0];
  float e[8], s = 0.f;
#pragma unroll
  for (int k = 0; k < 8; ++k) { e[k] = expf(val[k] - mx); s += e[k]; }
#pragma unroll
  for (int i = 0; i < NN_; ++i) p[i] = 0.f;
#pragma unroll
  for (int k = 0; k < 8; ++k) p[idx[k]] = e[k] / s;
}

// ---------------------------------------------------------------------------
// Persistent recurrence kernel: 32 blocks x 256 threads (8 waves).
// Block b owns channels [b*32, b*32+32). Weight slices pinned in LDS (256KB),
// staged by the Tensor Data Mover when available.
// ---------------------------------------------------------------------------
__global__ void __launch_bounds__(256, 1) recurrence_kernel(
    const float* __restrict__ prein, const float* __restrict__ taud,
    const float* __restrict__ gate,
    const bf16_t* __restrict__ Wrec, const bf16_t* __restrict__ Wk,
    const bf16_t* __restrict__ Wv,   const bf16_t* __restrict__ Wrg,
    const float* __restrict__ mu_k,  const float* __restrict__ mu_v,
    const float* __restrict__ mu_r,  const float* __restrict__ w_decay,
    const float* __restrict__ u_bonus,
    bf16_t* __restrict__ hbf, bf16_t* __restrict__ xkb,
    bf16_t* __restrict__ xvb, bf16_t* __restrict__ xrb,
    bf16_t* __restrict__ Ybf, unsigned* __restrict__ ctr) {
  extern __shared__ char smem[];
  bf16_t* Wls = (bf16_t*)smem;                                   // 4 x [32][1024]
  float*  red = (float*)(smem + 4 * CPB_ * D_ * sizeof(bf16_t)); // [8][16][16]

  const int tid  = threadIdx.x;
  const int lane = tid & 31;
  const int wave = tid >> 5;
  const int c0   = blockIdx.x * CPB_;

  const bf16_t* gsrc[4] = { Wrec + (size_t)c0 * D_, Wk + (size_t)c0 * D_,
                            Wv + (size_t)c0 * D_,   Wrg + (size_t)c0 * D_ };
#if HAVE_TDM_
  if (wave == 0) {
    // One TDM descriptor per 32x1024 bf16 weight slice (64KB each).
    for (int mtx = 0; mtx < 4; ++mtx) {
      bf16_t* dst = Wls + (size_t)mtx * CPB_ * D_;
      const unsigned lds_addr = (unsigned)(size_t)(const void*)dst;
      tdm_load_tile_2d(lds_addr, gsrc[mtx], CPB_, D_, D_);
    }
    __builtin_amdgcn_s_wait_tensorcnt(0);
  }
#else
  {
    const int vecs = CPB_ * D_ * (int)sizeof(bf16_t) / 16;  // 4096 x 16B
    for (int mtx = 0; mtx < 4; ++mtx) {
      const uint4* s = (const uint4*)gsrc[mtx];
      uint4* d = (uint4*)(Wls + (size_t)mtx * CPB_ * D_);
      for (int i = tid; i < vecs; i += 256) d[i] = s[i];
    }
  }
#endif
  __syncthreads();

  // Each thread permanently owns (batch bq, channel c0+c): state in registers.
  const int bq = tid >> 5;        // 0..7
  const int c  = tid & 31;        // 0..31
  const int ch = c0 + c;
  const float muk = mu_k[ch], muv = mu_v[ch], mur = mu_r[ch];
  const float wdec = -expf(w_decay[ch]);
  const float ub = u_bonus[ch];
  float h = 0.f, aa = 0.f, bbv = 0.f, pp = -1e38f;

  for (int t = 0; t < T_; ++t) {
    const int pbuf = t & 1, nbuf = pbuf ^ 1;

    // ---- phase 1: rec = h @ W_rec^T, split-K: wave -> (ntile, kslice) ----
    {
      const bf16_t* Ab = hbf + pbuf * (16 * D_);
      const int nt = wave & 1, ks = wave >> 1;
      v8f acc = {0.f, 0.f, 0.f, 0.f, 0.f, 0.f, 0.f, 0.f};
#pragma unroll
      for (int kk = 0; kk < 8; ++kk) {
        const int k0 = ks * 256 + kk * 32;
        v16bf a = load_a_frag(Ab, D_, 0, k0, lane);
        v16bf b = load_b_frag(Wls, D_, nt * 16, k0, lane);
        acc = __builtin_amdgcn_wmma_f32_16x16x32_bf16(false, a, false, b,
                                                      (short)0, acc, false, false);
      }
      float* dst = red + wave * 256;
#pragma unroll
      for (int j = 0; j < 8; ++j) {
        const int m = j + ((lane & 16) ? 8 : 0);
        dst[m * 16 + (lane & 15)] = acc[j];
      }
    }
    __syncthreads();

    // ---- phase 1 elementwise: liquid ODE update, publish h/xk/xv/xr ----
    {
      const int nt = c >> 4, nn = c & 15;
      const float rec = red[(0 * 2 + nt) * 256 + bq * 16 + nn]
                      + red[(1 * 2 + nt) * 256 + bq * 16 + nn]
                      + red[(2 * 2 + nt) * 256 + bq * 16 + nn]
                      + red[(3 * 2 + nt) * 256 + bq * 16 + nn];
      const size_t row = (size_t)bq * T_ + t;
      const size_t idx = row * D_ + ch;
      __builtin_prefetch(&prein[idx + D_], 0, 1);   // next-step rows
      __builtin_prefetch(&taud[idx + D_], 0, 1);
      const float act = tanhf(prein[idx] + rec);
      const float hold = h;
      h = hold + DT_ * act - taud[idx] * hold;      // taud = DT/tau
      const float xk = muk * h + (1.f - muk) * hold;  // h_prev == h at entry
      const float xv = muv * h + (1.f - muv) * hold;
      const float xr = mur * h + (1.f - mur) * hold;
      const int bi = bq * D_ + ch;
      hbf[nbuf * 16 * D_ + bi] = (bf16_t)h;
      xkb[pbuf * 16 * D_ + bi] = (bf16_t)xk;
      xvb[pbuf * 16 * D_ + bi] = (bf16_t)xv;
      xrb[pbuf * 16 * D_ + bi] = (bf16_t)xr;
    }
    __threadfence();
    __syncthreads();
    if (tid == 0) {
      atomicAdd(ctr, 1u);
      const unsigned tgt = (unsigned)NBLK_ * (unsigned)(t + 1);
      while (*(volatile unsigned*)ctr < tgt) __builtin_amdgcn_s_sleep(1);
    }
    __syncthreads();
    __threadfence();

    // ---- phase 2: k/v/r GEMMs: wave(0..5) -> (gemm, ntile), full K ----
    if (wave < 6) {
      const int g = wave >> 1, nt = wave & 1;
      const bf16_t* Ab = (g == 0 ? xkb : (g == 1 ? xvb : xrb)) + pbuf * 16 * D_;
      const bf16_t* Ws = Wls + (size_t)(g + 1) * CPB_ * D_;
      v8f acc = {0.f, 0.f, 0.f, 0.f, 0.f, 0.f, 0.f, 0.f};
      for (int kt = 0; kt < 32; ++kt) {
        const int k0 = kt * 32;
        v16bf a = load_a_frag(Ab, D_, 0, k0, lane);
        v16bf b = load_b_frag(Ws, D_, nt * 16, k0, lane);
        acc = __builtin_amdgcn_wmma_f32_16x16x32_bf16(false, a, false, b,
                                                      (short)0, acc, false, false);
      }
      float* dst = red + wave * 256;
#pragma unroll
      for (int j = 0; j < 8; ++j) {
        const int m = j + ((lane & 16) ? 8 : 0);
        dst[m * 16 + (lane & 15)] = acc[j];
      }
    }
    __syncthreads();

    // ---- phase 2 elementwise: WKV recurrence, spike, gate, emit y ----
    {
      const int nt = c >> 4, nn = c & 15;
      const float kval = red[(0 + nt) * 256 + bq * 16 + nn];
      const float vval = red[(2 + nt) * 256 + bq * 16 + nn];
      const float rpre = red[(4 + nt) * 256 + bq * 16 + nn];
      const float r = sigm_(rpre);
      const float ww = ub + kval;
      const float p  = fmaxf(pp, ww);
      const float e1 = expf(pp - p), e2 = expf(ww - p);
      const float wkv = (e1 * aa + e2 * vval) / (e1 * bbv + e2);
      const float ww2 = pp + wdec;
      const float p2  = fmaxf(ww2, kval);
      const float e1b = expf(ww2 - p2), e2b = expf(kval - p2);
      aa = e1b * aa + e2b * vval;
      bbv = e1b * bbv + e2b;
      pp = p2;
      const float spike = sigm_(SLOPE_ * (wkv - THR_));
      const size_t row = (size_t)bq * T_ + t;
      const float gt = gate[row * NN_ + (ch >> 4)];
      Ybf[row * D_ + ch] = (bf16_t)(r * wkv * spike * gt);
    }
    __syncthreads();   // red[] reused by next step's phase 1
  }
}

// ---------------------------------------------------------------------------
// Host-side orchestration (all launches on `stream`, graph-capture safe)
// ---------------------------------------------------------------------------
extern "C" void kernel_launch(void* const* d_in, const int* in_sizes, int n_in,
                              void* d_out, int out_size, void* d_ws, size_t ws_size,
                              hipStream_t stream) {
  (void)in_sizes; (void)n_in; (void)out_size; (void)ws_size;

  const float* x       = (const float*)d_in[0];
  const float* W_rec   = (const float*)d_in[1];
  const float* W_in    = (const float*)d_in[2];
  const float* W_tau   = (const float*)d_in[3];
  const float* bias    = (const float*)d_in[4];
  const float* tau_b   = (const float*)d_in[5];
  const float* mu_k    = (const float*)d_in[6];
  const float* mu_v    = (const float*)d_in[7];
  const float* mu_r    = (const float*)d_in[8];
  const float* Wk      = (const float*)d_in[9];
  const float* Wv      = (const float*)d_in[10];
  const float* Wr_g    = (const float*)d_in[11];
  const float* w_decay = (const float*)d_in[12];
  const float* u_bonus = (const float*)d_in[13];
  const float* routerW = (const float*)d_in[14];
  const float* routerB = (const float*)d_in[15];
  const float* W_out   = (const float*)d_in[16];
  float* out = (float*)d_out;

  char* ws = (char*)d_ws;
  size_t off = 0;
  auto take = [&](size_t bytes) -> char* {
    char* p = ws + off;
    off = (off + bytes + 255) & ~(size_t)255;
    return p;
  };
  bf16_t* Xbf   = (bf16_t*)take((size_t)M_ * D_ * 2);
  bf16_t* Wrecb = (bf16_t*)take((size_t)D_ * D_ * 2);
  bf16_t* Winb  = (bf16_t*)take((size_t)D_ * D_ * 2);
  bf16_t* Wtaub = (bf16_t*)take((size_t)D_ * D_ * 2);
  bf16_t* Wkb   = (bf16_t*)take((size_t)D_ * D_ * 2);
  bf16_t* Wvb   = (bf16_t*)take((size_t)D_ * D_ * 2);
  bf16_t* Wrgb  = (bf16_t*)take((size_t)D_ * D_ * 2);
  bf16_t* Woutb = (bf16_t*)take((size_t)D_ * D_ * 2);
  bf16_t* Wrtb  = (bf16_t*)take((size_t)NN_ * D_ * 2);
  float*  taud  = (float*)take((size_t)M_ * D_ * 4);   // DT/tau
  float*  prein = (float*)take((size_t)M_ * D_ * 4);   // x@W_in^T + bias
  float*  gateb = (float*)take((size_t)M_ * NN_ * 4);  // logits -> gates
  bf16_t* Ybf   = (bf16_t*)take((size_t)M_ * D_ * 2);
  // Contiguous zero-init region: state ping-pong buffers + barrier counter
  bf16_t* hbf   = (bf16_t*)take((size_t)2 * 16 * D_ * 2);
  bf16_t* xkb   = (bf16_t*)take((size_t)2 * 16 * D_ * 2);
  bf16_t* xvb   = (bf16_t*)take((size_t)2 * 16 * D_ * 2);
  bf16_t* xrb   = (bf16_t*)take((size_t)2 * 16 * D_ * 2);
  unsigned* ctr = (unsigned*)take(256);

  // f32 -> bf16 conversions
  auto cvt = [&](const float* s, bf16_t* d, int n) {
    cvt_kernel<<<dim3((n + 255) / 256), dim3(256), 0, stream>>>(s, d, n);
  };
  cvt(x, Xbf, M_ * D_);
  cvt(W_rec, Wrecb, D_ * D_);
  cvt(W_in, Winb, D_ * D_);
  cvt(W_tau, Wtaub, D_ * D_);
  cvt(Wk, Wkb, D_ * D_);
  cvt(Wv, Wvb, D_ * D_);
  cvt(Wr_g, Wrgb, D_ * D_);
  cvt(W_out, Woutb, D_ * D_);
  cvt(routerW, Wrtb, NN_ * D_);

  // Zero state buffers + barrier counter (contiguous, 256-aligned sizes)
  const int zwords = (4 * (2 * 16 * D_ * 2) + 256) / 4;
  zero_kernel<<<dim3((zwords + 255) / 256), dim3(256), 0, stream>>>((unsigned*)hbf, zwords);

  // Parallel precompute: tau, pre_in, router logits (batched over B*T rows)
  wmma_gemm_bf16<<<dim3(M_ / 16, D_ / 128), dim3(128), 0, stream>>>(Xbf, Wtaub, tau_b, taud, D_, 0);
  wmma_gemm_bf16<<<dim3(M_ / 16, D_ / 128), dim3(128), 0, stream>>>(Xbf, Winb, bias, prein, D_, 1);
  wmma_gemm_bf16<<<dim3(M_ / 16, 1), dim3(64), 0, stream>>>(Xbf, Wrtb, routerB, gateb, NN_, 1);
  topk_gate_kernel<<<dim3(M_ / 256), dim3(256), 0, stream>>>(gateb, M_);

  // Serial core: persistent recurrence, weights pinned in LDS (270,336 B)
  const size_t smem = (size_t)4 * CPB_ * D_ * sizeof(bf16_t) + (size_t)8 * 256 * sizeof(float);
  recurrence_kernel<<<dim3(NBLK_), dim3(256), smem, stream>>>(
      prein, taud, gateb, Wrecb, Wkb, Wvb, Wrgb,
      mu_k, mu_v, mu_r, w_decay, u_bonus,
      hbf, xkb, xvb, xrb, Ybf, ctr);

  // Deferred output projection: out = Y @ W_out^T (batched)
  wmma_gemm_bf16<<<dim3(M_ / 16, D_ / 128), dim3(128), 0, stream>>>(Ybf, Woutb, nullptr, out, D_, 2);
}
